// MultiHeadAttention_38697655337195
// MI455X (gfx1250) — matrix operand order repaired and checked by reading, not compiled, not run on hardware
//
#include <hip/hip_runtime.h>
#include <hip/hip_bf16.h>
#include <stdint.h>

typedef __attribute__((ext_vector_type(16))) _Float16 v16h;
typedef __attribute__((ext_vector_type(8)))  _Float16 v8h;
typedef __attribute__((ext_vector_type(8)))  float    v8f;

#define B_   4
#define S_   1024
#define D_   1024
#define H_   16
#define DK_  64
#define M_   128
#define SKM_ (S_ + M_)   // 1152

// ---------------------------------------------------------------------------
// CDNA5 async memory->LDS copy (ASYNCcnt-tracked). 16 bytes per lane.
//   LDS destination at offset lds_off; global source at base + voff bytes
// ---------------------------------------------------------------------------
static __device__ __forceinline__ void async_copy_b128(uint32_t lds_off, uint32_t voff,
                                                       const void* base) {
  asm volatile("global_load_async_to_lds_b128 %0, %1, %2"
               :: "v"(lds_off), "v"(voff), "s"(base)
               : "memory");
}
#define WAIT_ASYNCCNT0() asm volatile("s_wait_asynccnt 0" ::: "memory")

static __device__ __forceinline__ uint32_t lds_addr(const void* p) {
  return (uint32_t)(uintptr_t)p;   // low 32 bits of generic LDS pointer = LDS offset
}

// ---------------------------------------------------------------------------
// WMMA helpers (CDNA5 16x16x32 f16 -> f32)
// ---------------------------------------------------------------------------
static __device__ __forceinline__ v8f wmma_f16(v16h a, v16h b, v8f c) {
  return __builtin_amdgcn_wmma_f32_16x16x32_f16(
      /*neg_a=*/false, a, /*neg_b=*/false, b,
      /*c_mod=*/(short)0, c, /*reuse_a=*/false, /*reuse_b=*/false);
}

// A-matrix 16x32 f16 fragment (row-major source, stride in halves).
// Lanes 0-15 -> M=lane, K {0..7,16..23}; lanes 16-31 -> K {8..15,24..31}.
static __device__ __forceinline__ v16h load_a_frag(const _Float16* base, int stride, int lane) {
  const int r  = lane & 15;
  const int kg = (lane >> 4) * 8;
  const _Float16* p = base + r * stride + kg;
  v8h lo = *(const v8h*)(p);
  v8h hi = *(const v8h*)(p + 16);
  v16h f;
#pragma unroll
  for (int i = 0; i < 8; ++i) { f[i] = lo[i]; f[i + 8] = hi[i]; }
  return f;
}

// B-matrix 32x16 f16 fragment from N-major buffer baseT[n*stride + k].
// Lanes 0-15 -> N=lane, K {0..7,16..23}; lanes 16-31 -> K {8..15,24..31}.
static __device__ __forceinline__ v16h load_b_frag(const _Float16* baseT, int stride, int lane) {
  const int n  = lane & 15;
  const int k0 = (lane >> 4) * 8;
  const _Float16* p = baseT + n * stride + k0;
  v8h lo = *(const v8h*)(p);
  v8h hi = *(const v8h*)(p + 16);
  v16h f;
#pragma unroll
  for (int i = 0; i < 8; ++i) { f[i] = lo[i]; f[i + 8] = hi[i]; }
  return f;
}

// ---------------------------------------------------------------------------
// Prep kernels
// ---------------------------------------------------------------------------
__global__ __launch_bounds__(256) void cast_f16_kernel(const float* __restrict__ in,
                                                       _Float16* __restrict__ out, int n) {
  int i = blockIdx.x * 256 + threadIdx.x;
  if (i < n) out[i] = (_Float16)in[i];
}

// W [K=1024][N=1024] f32 -> Wt [N][K] f16, LDS-tiled (coalesced both sides)
__global__ __launch_bounds__(256) void transpose_w_kernel(const float* __restrict__ in,
                                                          _Float16* __restrict__ out) {
  __shared__ float tile[32][33];
  const int tx = threadIdx.x & 31;
  const int ty = threadIdx.x >> 5;          // 0..7
  const int kb = blockIdx.x * 32;           // K block
  const int nb = blockIdx.y * 32;           // N block
#pragma unroll
  for (int i = 0; i < 4; ++i)
    tile[ty + 8 * i][tx] = in[(size_t)(kb + ty + 8 * i) * D_ + nb + tx];
  __syncthreads();
#pragma unroll
  for (int i = 0; i < 4; ++i)
    out[(size_t)(nb + ty + 8 * i) * D_ + kb + tx] = (_Float16)tile[tx][ty + 8 * i];
}

// Memory slots: Kall[b][S_+m][c] = sqrt(M)*mk[m][c]
__global__ __launch_bounds__(256) void fill_memk_kernel(const float* __restrict__ mk,
                                                        _Float16* __restrict__ Kall) {
  int i = blockIdx.x * 256 + threadIdx.x;
  int c = i & (D_ - 1);
  int t = i >> 10;
  int m = t & (M_ - 1);
  int b = t >> 7;
  float v = 11.31370849898476f * mk[(size_t)m * D_ + c];
  Kall[((size_t)(b * SKM_ + S_ + m)) * D_ + c] = (_Float16)v;
}

// Vt[b][c][S_+m] = sqrt(M)*mv[m][c]
__global__ __launch_bounds__(256) void fill_memv_kernel(const float* __restrict__ mv,
                                                        _Float16* __restrict__ Vt) {
  int i = blockIdx.x * 256 + threadIdx.x;
  int c = i & (D_ - 1);
  int t = i >> 10;
  int m = t & (M_ - 1);
  int b = t >> 7;
  float v = 11.31370849898476f * mv[(size_t)m * D_ + c];
  Vt[((size_t)(b * D_ + c)) * SKM_ + (S_ + m)] = (_Float16)v;
}

// ---------------------------------------------------------------------------
// Tiled WMMA GEMM: C[M x 1024] = A[M x 1024](f16) @ Bt[1024 x 1024](f16, N-major) + bias
// Block 128x128, 8 waves (4x2), wave tile 32x64, K-step 32.
// Double-buffered LDS filled by async DMA (ASYNCcnt), one barrier per K-step.
// OUT_MODE: 0=f32 row-major; 1=f16 row-major scaled (Q); 2=f16 K_all remap; 3=f16 Vt.
// ---------------------------------------------------------------------------
template <int OUT_MODE>
__global__ __launch_bounds__(256) void gemm_kernel(const _Float16* __restrict__ A,
                                                   const _Float16* __restrict__ Bt,
                                                   const float* __restrict__ bias,
                                                   void* __restrict__ out, float scale) {
  __shared__ _Float16 As[2][128][48];
  __shared__ _Float16 Bs[2][128][48];

  const int tid  = threadIdx.x;
  const int lane = tid & 31;
  const int w    = tid >> 5;
  const int wm   = w >> 1;           // 0..3
  const int wn   = w & 1;            // 0..1
  const int rowBase = blockIdx.y * 128;
  const int colBase = blockIdx.x * 128;

  const _Float16* baseA = A  + (size_t)rowBase * D_;
  const _Float16* baseB = Bt + (size_t)colBase * D_;

  v8f acc[2][4];
#pragma unroll
  for (int i = 0; i < 2; ++i)
#pragma unroll
    for (int j = 0; j < 4; ++j) acc[i][j] = v8f{};

  const int sr = tid >> 2;           // 0..63
  const int sc = (tid & 3) * 8;      // halves

  auto stage = [&](int k0, int buf) {
    uint32_t va0 = (uint32_t)(((sr)      * D_ + k0 + sc) * 2);
    uint32_t va1 = (uint32_t)(((sr + 64) * D_ + k0 + sc) * 2);
    async_copy_b128(lds_addr(&As[buf][sr][sc]),      va0, baseA);
    async_copy_b128(lds_addr(&As[buf][sr + 64][sc]), va1, baseA);
    async_copy_b128(lds_addr(&Bs[buf][sr][sc]),      va0, baseB);
    async_copy_b128(lds_addr(&Bs[buf][sr + 64][sc]), va1, baseB);
  };

  stage(0, 0);
  WAIT_ASYNCCNT0();
  __syncthreads();

  const int nIter = D_ / 32;
  for (int it = 0; it < nIter; ++it) {
    const int buf = it & 1;
    if (it + 1 < nIter) stage((it + 1) * 32, buf ^ 1);

    v16h a0 = load_a_frag(&As[buf][wm * 32 +  0][0], 48, lane);
    v16h a1 = load_a_frag(&As[buf][wm * 32 + 16][0], 48, lane);
    v16h b0 = load_b_frag(&Bs[buf][wn * 64 +  0][0], 48, lane);
    v16h b1 = load_b_frag(&Bs[buf][wn * 64 + 16][0], 48, lane);
    v16h b2 = load_b_frag(&Bs[buf][wn * 64 + 32][0], 48, lane);
    v16h b3 = load_b_frag(&Bs[buf][wn * 64 + 48][0], 48, lane);

    acc[0][0] = wmma_f16(a0, b0, acc[0][0]);
    acc[0][1] = wmma_f16(a0, b1, acc[0][1]);
    acc[0][2] = wmma_f16(a0, b2, acc[0][2]);
    acc[0][3] = wmma_f16(a0, b3, acc[0][3]);
    acc[1][0] = wmma_f16(a1, b0, acc[1][0]);
    acc[1][1] = wmma_f16(a1, b1, acc[1][1]);
    acc[1][2] = wmma_f16(a1, b2, acc[1][2]);
    acc[1][3] = wmma_f16(a1, b3, acc[1][3]);

    WAIT_ASYNCCNT0();
    __syncthreads();
  }

  // Epilogue. C layout: lanes 0-15 -> N=lane, M=r; lanes 16-31 -> N=lane-16, M=r+8.
  const int cl = lane & 15;
  const int rh = (lane >> 4) * 8;
#pragma unroll
  for (int mi = 0; mi < 2; ++mi) {
#pragma unroll
    for (int ni = 0; ni < 4; ++ni) {
      const int col = colBase + wn * 64 + ni * 16 + cl;
      const float bs = bias[col];
#pragma unroll
      for (int r = 0; r < 8; ++r) {
        const int row = rowBase + wm * 32 + mi * 16 + rh + r;
        const float v = (acc[mi][ni][r] + bs) * scale;
        if constexpr (OUT_MODE == 0) {
          ((float*)out)[(size_t)row * D_ + col] = v;
        } else if constexpr (OUT_MODE == 1) {
          ((_Float16*)out)[(size_t)row * D_ + col] = (_Float16)v;
        } else if constexpr (OUT_MODE == 2) {
          const int orow = row + (row >> 10) * M_;   // b*SKM_ + s
          ((_Float16*)out)[(size_t)orow * D_ + col] = (_Float16)v;
        } else {  // 3: Vt[b][col][s]
          const int b = row >> 10;
          const int s = row & (S_ - 1);
          ((_Float16*)out)[((size_t)(b * D_ + col)) * SKM_ + s] = (_Float16)v;
        }
      }
    }
  }
}

// ---------------------------------------------------------------------------
// Flash attention: grid (S/128, B*H), 256 threads = 8 waves, wave owns 16 q rows.
// Q pre-scaled by 1/sqrt(dk). 64-key chunks, double-buffered async staging.
// ---------------------------------------------------------------------------
__global__ __launch_bounds__(256) void attn_kernel(const _Float16* __restrict__ Q,
                                                   const _Float16* __restrict__ Kall,
                                                   const _Float16* __restrict__ Vt,
                                                   _Float16* __restrict__ O) {
  __shared__ _Float16 Ks[2][64][72];     // [key][d]
  __shared__ _Float16 Vs[2][64][72];     // [d][key]
  __shared__ _Float16 Ps[8][16][72];     // per-wave P tile [qrow][key]

  const int tid  = threadIdx.x;
  const int lane = tid & 31;
  const int w    = tid >> 5;
  const int bh   = blockIdx.y;
  const int b    = bh >> 4;
  const int h    = bh & 15;
  const int qbase = blockIdx.x * 128 + w * 16;

  const _Float16* qptr  = Q + ((size_t)(b * S_ + qbase)) * D_ + h * DK_;
  const _Float16* kbase = Kall + (size_t)b * SKM_ * D_ + h * DK_;
  const _Float16* vbase = Vt + ((size_t)(b * D_ + h * DK_)) * SKM_;

  const v16h aq0 = load_a_frag(qptr,      D_, lane);
  const v16h aq1 = load_a_frag(qptr + 32, D_, lane);

  v8f o[4];
#pragma unroll
  for (int f = 0; f < 4; ++f) o[f] = v8f{};
  float mrow[8], lrow[8];
#pragma unroll
  for (int r = 0; r < 8; ++r) { mrow[r] = -1e30f; lrow[r] = 0.0f; }

  const int cl = lane & 15;
  const int rh = (lane >> 4) * 8;
  const int sr = tid >> 2;           // 0..63
  const int sc = (tid & 3) * 8;      // 0..24 halves

  auto stage = [&](int kcb, int buf) {
    uint32_t kv = (uint32_t)(((kcb + sr) * D_ + sc) * 2);
    async_copy_b128(lds_addr(&Ks[buf][sr][sc]),      kv,      kbase);
    async_copy_b128(lds_addr(&Ks[buf][sr][sc + 32]), kv + 64, kbase);
    uint32_t vv = (uint32_t)((sr * SKM_ + kcb + sc) * 2);
    async_copy_b128(lds_addr(&Vs[buf][sr][sc]),      vv,      vbase);
    async_copy_b128(lds_addr(&Vs[buf][sr][sc + 32]), vv + 64, vbase);
  };

  stage(0, 0);
  WAIT_ASYNCCNT0();
  __syncthreads();

  const int nIter = SKM_ / 64;       // 18
  for (int it = 0; it < nIter; ++it) {
    const int buf = it & 1;
    if (it + 1 < nIter) stage((it + 1) * 64, buf ^ 1);

    const _Float16(*ks)[72] = Ks[buf];
    const _Float16(*vs)[72] = Vs[buf];

    // Scores: 16 q x 64 keys = 4 fragments, 2 WMMAs each (dk = 64).
    v8f s[4];
#pragma unroll
    for (int j = 0; j < 4; ++j) {
      v16h kb0 = load_b_frag(&ks[j * 16][0],  72, lane);
      v16h kb1 = load_b_frag(&ks[j * 16][32], 72, lane);
      s[j] = v8f{};
      s[j] = wmma_f16(aq0, kb0, s[j]);
      s[j] = wmma_f16(aq1, kb1, s[j]);
    }

    // Online softmax: row m lives in VGPR slot r across a 16-lane column group.
    float p[4][8], corr[8];
#pragma unroll
    for (int r = 0; r < 8; ++r) {
      float cm = fmaxf(fmaxf(s[0][r], s[1][r]), fmaxf(s[2][r], s[3][r]));
      cm = fmaxf(cm, __shfl_xor(cm, 1, 32));
      cm = fmaxf(cm, __shfl_xor(cm, 2, 32));
      cm = fmaxf(cm, __shfl_xor(cm, 4, 32));
      cm = fmaxf(cm, __shfl_xor(cm, 8, 32));
      const float nm = fmaxf(mrow[r], cm);
      corr[r] = __expf(mrow[r] - nm);
      float cs = 0.0f;
#pragma unroll
      for (int j = 0; j < 4; ++j) {
        p[j][r] = __expf(s[j][r] - nm);
        cs += p[j][r];
      }
      cs += __shfl_xor(cs, 1, 32);
      cs += __shfl_xor(cs, 2, 32);
      cs += __shfl_xor(cs, 4, 32);
      cs += __shfl_xor(cs, 8, 32);
      lrow[r] = lrow[r] * corr[r] + cs;
      mrow[r] = nm;
    }
#pragma unroll
    for (int f = 0; f < 4; ++f)
#pragma unroll
      for (int r = 0; r < 8; ++r) o[f][r] *= corr[r];

    // P (C layout) -> per-wave LDS -> A layout fragments (keys 0..31, 32..63).
#pragma unroll
    for (int j = 0; j < 4; ++j)
#pragma unroll
      for (int r = 0; r < 8; ++r)
        Ps[w][rh + r][j * 16 + cl] = (_Float16)p[j][r];

    v16h pa0 = load_a_frag(&Ps[w][0][0],  72, lane);
    v16h pa1 = load_a_frag(&Ps[w][0][32], 72, lane);
#pragma unroll
    for (int f = 0; f < 4; ++f) {
      v16h vb0 = load_b_frag(&vs[f * 16][0],  72, lane);
      v16h vb1 = load_b_frag(&vs[f * 16][32], 72, lane);
      o[f] = wmma_f16(pa0, vb0, o[f]);
      o[f] = wmma_f16(pa1, vb1, o[f]);
    }

    WAIT_ASYNCCNT0();
    __syncthreads();
  }

  // Normalize and write f16 output [B*S][1024] at column block h*64.
  float inv[8];
#pragma unroll
  for (int r = 0; r < 8; ++r) inv[r] = 1.0f / lrow[r];
#pragma unroll
  for (int f = 0; f < 4; ++f) {
#pragma unroll
    for (int r = 0; r < 8; ++r) {
      const int row = b * S_ + qbase + rh + r;
      const int col = h * DK_ + f * 16 + cl;
      O[(size_t)row * D_ + col] = (_Float16)(o[f][r] * inv[r]);
    }
  }
}

// ---------------------------------------------------------------------------
// Host launcher
// ---------------------------------------------------------------------------
extern "C" void kernel_launch(void* const* d_in, const int* in_sizes, int n_in,
                              void* d_out, int out_size, void* d_ws, size_t ws_size,
                              hipStream_t stream) {
  (void)in_sizes; (void)n_in; (void)out_size; (void)ws_size;
  const float* queries = (const float*)d_in[0];
  const float* keys    = (const float*)d_in[1];
  const float* values  = (const float*)d_in[2];
  const float* Wq = (const float*)d_in[3];
  const float* bq = (const float*)d_in[4];
  const float* Wk = (const float*)d_in[5];
  const float* bk = (const float*)d_in[6];
  const float* Wv = (const float*)d_in[7];
  const float* bv = (const float*)d_in[8];
  const float* Wo = (const float*)d_in[9];
  const float* bo = (const float*)d_in[10];
  const float* mk = (const float*)d_in[11];
  const float* mv = (const float*)d_in[12];
  float* out = (float*)d_out;

  char* ws = (char*)d_ws;
  size_t off = 0;
  auto alloc = [&](size_t bytes) -> void* {
    void* p = ws + off;
    off += (bytes + 255) & ~(size_t)255;
    return p;
  };
  const size_t NT = (size_t)B_ * S_;            // 4096 tokens
  _Float16* xq   = (_Float16*)alloc(NT * D_ * 2);
  _Float16* xk   = (_Float16*)alloc(NT * D_ * 2);
  _Float16* xv   = (_Float16*)alloc(NT * D_ * 2);
  _Float16* wqt  = (_Float16*)alloc((size_t)D_ * D_ * 2);
  _Float16* wkt  = (_Float16*)alloc((size_t)D_ * D_ * 2);
  _Float16* wvt  = (_Float16*)alloc((size_t)D_ * D_ * 2);
  _Float16* wot  = (_Float16*)alloc((size_t)D_ * D_ * 2);
  _Float16* qf   = (_Float16*)alloc(NT * D_ * 2);
  _Float16* kall = (_Float16*)alloc((size_t)B_ * SKM_ * D_ * 2);
  _Float16* vt   = (_Float16*)alloc((size_t)B_ * D_ * SKM_ * 2);
  _Float16* attn = (_Float16*)alloc(NT * D_ * 2);

  const int nElem = (int)(NT * D_);             // 4194304
  cast_f16_kernel<<<nElem / 256, 256, 0, stream>>>(queries, xq, nElem);
  cast_f16_kernel<<<nElem / 256, 256, 0, stream>>>(keys,    xk, nElem);
  cast_f16_kernel<<<nElem / 256, 256, 0, stream>>>(values,  xv, nElem);

  dim3 tg(D_ / 32, D_ / 32);
  transpose_w_kernel<<<tg, 256, 0, stream>>>(Wq, wqt);
  transpose_w_kernel<<<tg, 256, 0, stream>>>(Wk, wkt);
  transpose_w_kernel<<<tg, 256, 0, stream>>>(Wv, wvt);
  transpose_w_kernel<<<tg, 256, 0, stream>>>(Wo, wot);

  const int nMem = B_ * M_ * D_;                // 524288
  fill_memk_kernel<<<nMem / 256, 256, 0, stream>>>(mk, kall);
  fill_memv_kernel<<<nMem / 256, 256, 0, stream>>>(mv, vt);

  dim3 gg(D_ / 128, (int)(NT / 128));           // (8, 32)
  gemm_kernel<1><<<gg, 256, 0, stream>>>(xq, wqt, bq, qf,   0.125f);  // Q / sqrt(dk)
  gemm_kernel<2><<<gg, 256, 0, stream>>>(xk, wkt, bk, kall, 1.0f);
  gemm_kernel<3><<<gg, 256, 0, stream>>>(xv, wvt, bv, vt,   1.0f);

  dim3 ga(S_ / 128, B_ * H_);                   // (8, 64)
  attn_kernel<<<ga, 256, 0, stream>>>(qf, kall, vt, attn);

  gemm_kernel<0><<<gg, 256, 0, stream>>>(attn, wot, bo, out, 1.0f);
}
